// Chebyshev_KAN_83322365542860
// MI455X (gfx1250) — matrix operand order repaired and checked by reading, not compiled, not run on hardware
//
#include <hip/hip_runtime.h>
#include <hip/hip_bf16.h>

// ---------------------------------------------------------------------------
// Chebyshev KAN (2 layers) as bf16 WMMA GEMMs on gfx1250.
//   layer(x, C):  xn = x*scale + bias (per-feature minmax -> [-1,1])
//                 y[b,o] = sum_{i,d} T_d(xn[b,i]) * C[i,o,d]
// Contract over K = (d,i): A generated on the fly via the Chebyshev
// recurrence evaluated in packed bf16 (the recurrence state IS the WMMA
// A-fragment), B = coeffs repacked to bf16 [d][o][i] (contiguous K runs).
// ---------------------------------------------------------------------------

typedef __bf16   bf16_t;
typedef bf16_t   v16bf __attribute__((ext_vector_type(16)));
typedef bf16_t   v2bf  __attribute__((ext_vector_type(2)));
typedef float    v8f   __attribute__((ext_vector_type(8)));
typedef float    v4f   __attribute__((ext_vector_type(4)));
typedef unsigned int v4u __attribute__((ext_vector_type(4)));

union FragB  { v16bf v; v4u q[2]; unsigned u[8]; unsigned short s[16]; };
union FragAcc{ v8f   v; float f[8]; };
union BfPair { v2bf  v; unsigned u; };

// Pack two f32 -> one dword of two bf16 (round-to-nearest; lo in low half).
__device__ __forceinline__ unsigned pack2bf(float lo, float hi) {
  unsigned ul = __builtin_bit_cast(unsigned, lo) + 0x8000u;
  unsigned uh = __builtin_bit_cast(unsigned, hi) + 0x8000u;
  // result bytes: [uh.b3, uh.b2, ul.b3, ul.b2]
  return __builtin_amdgcn_perm(uh, ul, 0x07060302u);
}

__device__ __forceinline__ unsigned short f2bf(float f) {
  unsigned u = __builtin_bit_cast(unsigned, f) + 0x8000u;
  return (unsigned short)(u >> 16);
}

// ---------------------------------------------------------------------------
// Per-feature min/max over the batch (two-pass), emits scale/bias:
//   xn = x*scale + bias  with scale = 2/(mx-mn), bias = -(mx+mn)/(mx-mn)
// ---------------------------------------------------------------------------
__global__ void minmax_pass1(const float* __restrict__ X, int I, int rowsPerChunk,
                             float* __restrict__ pmin, float* __restrict__ pmax) {
  int f = blockIdx.x * blockDim.x + threadIdx.x;
  if (f >= I) return;
  int c = blockIdx.y;
  const float* p = X + (size_t)c * rowsPerChunk * I + f;
  float mn = 3.402823466e38f, mx = -3.402823466e38f;
  for (int r = 0; r < rowsPerChunk; ++r) {
    float v = p[(size_t)r * I];
    mn = fminf(mn, v);
    mx = fmaxf(mx, v);
  }
  pmin[(size_t)c * I + f] = mn;
  pmax[(size_t)c * I + f] = mx;
}

__global__ void minmax_pass2(const float* __restrict__ pmin, const float* __restrict__ pmax,
                             int I, int nchunks,
                             float* __restrict__ scale, float* __restrict__ bias) {
  int f = blockIdx.x * blockDim.x + threadIdx.x;
  if (f >= I) return;
  float mn = 3.402823466e38f, mx = -3.402823466e38f;
  for (int c = 0; c < nchunks; ++c) {
    mn = fminf(mn, pmin[(size_t)c * I + f]);
    mx = fmaxf(mx, pmax[(size_t)c * I + f]);
  }
  float d = mx - mn;
  scale[f] = 2.0f / d;
  bias[f]  = -(mx + mn) / d;
}

// ---------------------------------------------------------------------------
// Repack coeffs: src [I, O, 9] f32  ->  dst [9, O, I] bf16
// ---------------------------------------------------------------------------
__global__ void convert_coeffs(const float* __restrict__ src, unsigned short* __restrict__ dst,
                               int I, int O) {
  int i = blockIdx.x * blockDim.x + threadIdx.x;
  int o = blockIdx.y;
  if (i >= I) return;
  const float* p = src + ((size_t)i * O + o) * 9;
  float f[9];
#pragma unroll
  for (int d = 0; d < 9; ++d) f[d] = p[d];
#pragma unroll
  for (int d = 0; d < 9; ++d)
    dst[((size_t)d * O + o) * I + i] = f2bf(f[d]);
}

// ---------------------------------------------------------------------------
// Fused Chebyshev expansion + GEMM with v_wmma_f32_16x16x32_bf16.
// Block = 128 threads = 4 waves. Wave w: 16-row band, 128 output cols
// (8 x 16x16 subtiles -> 64 accumulator VGPRs).
//
// A fragment (16-bit 16x32, ISA 7.12.2): lane M = lane&15, half = lane>>4,
//   packed halves j=0..7  -> K = half*8 + j
//   packed halves j=8..15 -> K = 16 + half*8 + (j-8)
// B fragment (32x16): lane N = lane&15, half = lane>>4,
//   packed halves j=0..15 -> same K as A   (two 16B runs at K = half*8, 16 + half*8)
// C/D (f32 16x16): lane n = lane&15, half; vgpr p -> row = half*8 + p.
// ---------------------------------------------------------------------------
#define TN 8   // 16x16 column subtiles per wave (128 cols)

__global__ __launch_bounds__(128)
void cheby_gemm_wmma(const float* __restrict__ X,        // [Bn, I] f32 (raw, pre-norm)
                     const float* __restrict__ scale,    // [I]
                     const float* __restrict__ bias,     // [I]
                     const unsigned short* __restrict__ Bt, // [9, O, I] bf16
                     float* __restrict__ C,              // [Bn, O] f32
                     int I, int O) {
  const int tid  = threadIdx.x;
  const int lane = tid & 31;
  const int wid  = tid >> 5;
  const int ln   = lane & 15;
  const int half = lane >> 4;

  const int r0 = blockIdx.y * 64 + wid * 16;    // row band of this wave
  const int o0 = blockIdx.x * (TN * 16);        // 128 output cols

  FragAcc acc[TN];
#pragma unroll
  for (int t = 0; t < TN; ++t)
#pragma unroll
    for (int p = 0; p < 8; ++p) acc[t].f[p] = 0.0f;

  const float* xrow = X + (size_t)(r0 + ln) * I;

  for (int i0 = 0; i0 < I; i0 += 32) {
    const int f1 = i0 + half * 8;        // features for packed j = 0..7
    const int f2 = i0 + 16 + half * 8;   // features for packed j = 8..15

    // Load x + per-feature scale/bias, normalize to [-1,1] (f32 precision).
    v4f xa = *(const v4f*)(xrow + f1);
    v4f xb = *(const v4f*)(xrow + f1 + 4);
    v4f xc = *(const v4f*)(xrow + f2);
    v4f xd = *(const v4f*)(xrow + f2 + 4);
    v4f sa = *(const v4f*)(scale + f1);
    v4f sb = *(const v4f*)(scale + f1 + 4);
    v4f sc = *(const v4f*)(scale + f2);
    v4f sd = *(const v4f*)(scale + f2 + 4);
    v4f ba = *(const v4f*)(bias + f1);
    v4f bb = *(const v4f*)(bias + f1 + 4);
    v4f bc = *(const v4f*)(bias + f2);
    v4f bd = *(const v4f*)(bias + f2 + 4);

    float xn[16];
#pragma unroll
    for (int j = 0; j < 4; ++j) {
      xn[j]      = xa[j] * sa[j] + ba[j];
      xn[4 + j]  = xb[j] * sb[j] + bb[j];
      xn[8 + j]  = xc[j] * sc[j] + bc[j];
      xn[12 + j] = xd[j] * sd[j] + bd[j];
    }

    // One-time per chunk: xn -> packed bf16, x2 = 2*xn in packed bf16.
    BfPair xnb[8], x2b[8], cur[8], prv[8];
#pragma unroll
    for (int jj = 0; jj < 8; ++jj) {
      xnb[jj].u = pack2bf(xn[2 * jj], xn[2 * jj + 1]);
      x2b[jj].v = xnb[jj].v + xnb[jj].v;      // v_pk_add_bf16
      cur[jj].u = 0x3F803F80u;                // T0 = {1.0bf16, 1.0bf16}
      prv[jj].u = 0u;
    }

#pragma unroll
    for (int d = 0; d < 9; ++d) {
      // A fragment is the bf16 recurrence state directly.
      FragB a;
#pragma unroll
      for (int jj = 0; jj < 8; ++jj) a.u[jj] = cur[jj].u;

      // B rows: Bt[d][o0 + t*16 + ln][i0 + half*8 .. +7] and [i0 + 16 + half*8 .. +7]
      const unsigned short* bbase =
          Bt + ((size_t)d * O + (size_t)(o0 + ln)) * I + i0 + half * 8;
#pragma unroll
      for (int t = 0; t < TN; ++t) {
        FragB b;
        const unsigned short* bp = bbase + (size_t)t * 16 * I;
        b.q[0] = *(const v4u*)(bp);
        b.q[1] = *(const v4u*)(bp + 16);
        acc[t].v = __builtin_amdgcn_wmma_f32_16x16x32_bf16(
            false, a.v, false, b.v, (short)0, acc[t].v, false, false);
      }

      // Recurrence in packed bf16: T_{d+1} = (2x)*T_d - T_{d-1}
#pragma unroll
      for (int jj = 0; jj < 8; ++jj) {
        v2bf nxt;
        if (d == 0) nxt = xnb[jj].v;                       // T1 = x
        else        nxt = x2b[jj].v * cur[jj].v - prv[jj].v; // v_pk_fma_bf16
        prv[jj].v = cur[jj].v;
        cur[jj].v = nxt;
      }
    }
  }

  // Store D tiles: row = r0 + half*8 + p, col = o0 + t*16 + ln.
#pragma unroll
  for (int t = 0; t < TN; ++t)
#pragma unroll
    for (int p = 0; p < 8; ++p)
      C[(size_t)(r0 + half * 8 + p) * O + (o0 + t * 16 + ln)] = acc[t].f[p];
}

// ---------------------------------------------------------------------------
extern "C" void kernel_launch(void* const* d_in, const int* in_sizes, int n_in,
                              void* d_out, int out_size, void* d_ws, size_t ws_size,
                              hipStream_t stream) {
  (void)in_sizes; (void)n_in; (void)out_size; (void)ws_size;

  constexpr int Bn = 16384;
  constexpr int I1 = 1024, O1 = 2048;
  constexpr int I2 = 2048, O2 = 1024;
  constexpr int NCHUNK = 32;

  const float* x  = (const float*)d_in[0];
  const float* c1 = (const float*)d_in[1];
  const float* c2 = (const float*)d_in[2];
  float* out = (float*)d_out;

  char* ws = (char*)d_ws;
  unsigned short* c1t = (unsigned short*)ws;                 // [9,O1,I1] bf16
  ws += (size_t)9 * O1 * I1 * sizeof(unsigned short);
  unsigned short* c2t = (unsigned short*)ws;                 // [9,O2,I2] bf16
  ws += (size_t)9 * O2 * I2 * sizeof(unsigned short);
  float* h = (float*)ws;                                     // [Bn,O1] f32
  ws += (size_t)Bn * O1 * sizeof(float);
  float* scale1 = (float*)ws; ws += I1 * sizeof(float);
  float* bias1  = (float*)ws; ws += I1 * sizeof(float);
  float* scale2 = (float*)ws; ws += I2 * sizeof(float);
  float* bias2  = (float*)ws; ws += I2 * sizeof(float);
  float* pmin   = (float*)ws; ws += (size_t)NCHUNK * I2 * sizeof(float);
  float* pmax   = (float*)ws; ws += (size_t)NCHUNK * I2 * sizeof(float);

  // Repack both coefficient tensors to bf16 [d][o][i].
  convert_coeffs<<<dim3(I1 / 256, O1), 256, 0, stream>>>(c1, c1t, I1, O1);
  convert_coeffs<<<dim3(I2 / 256, O2), 256, 0, stream>>>(c2, c2t, I2, O2);

  // Layer 1: minmax(x) -> scale/bias, then fused Chebyshev GEMM -> h.
  minmax_pass1<<<dim3(I1 / 256, NCHUNK), 256, 0, stream>>>(x, I1, Bn / NCHUNK, pmin, pmax);
  minmax_pass2<<<dim3(I1 / 256), 256, 0, stream>>>(pmin, pmax, I1, NCHUNK, scale1, bias1);
  cheby_gemm_wmma<<<dim3(O1 / (TN * 16), Bn / 64), 128, 0, stream>>>(x, scale1, bias1, c1t, h, I1, O1);

  // Layer 2: minmax(h) -> scale/bias, fused Chebyshev GEMM -> out.
  minmax_pass1<<<dim3(I2 / 256, NCHUNK), 256, 0, stream>>>(h, I2, Bn / NCHUNK, pmin, pmax);
  minmax_pass2<<<dim3(I2 / 256), 256, 0, stream>>>(pmin, pmax, I2, NCHUNK, scale2, bias2);
  cheby_gemm_wmma<<<dim3(O2 / (TN * 16), Bn / 64), 128, 0, stream>>>(h, scale2, bias2, c2t, out, I2, O2);
}
